// PointTransformerClassification_24988119728696
// MI455X (gfx1250) — matrix operand order repaired and checked by reading, not compiled, not run on hardware
//
#include <hip/hip_runtime.h>
#include <hip/hip_bf16.h>

// ---------------------------------------------------------------------------
// Point Transformer classification forward for MI455X (gfx1250, wave32).
// All C x C linear layers -> v_wmma_f32_16x16x32_f16 (f16 operands, f32 acc).
// Weights pre-packed to f16 zero-padded to K%32==0: WMMA inner loop issues
// only unconditional b128 loads. Each wave computes a 16x32 output slab
// (one shared A fragment, two B fragments, two WMMAs per K-step) to halve
// A-operand traffic on these skinny (huge-M, small-N) GEMMs.
// Tiny-K layers (K=3,4) use a plain VALU linear kernel.
// ---------------------------------------------------------------------------

typedef __attribute__((ext_vector_type(16))) _Float16 v16h;
typedef __attribute__((ext_vector_type(8)))  float    v8f;
typedef __attribute__((ext_vector_type(4)))  float    v4f;

#define THREADS 256
#define WAVES_PER_BLOCK 8

// ========================== WMMA GEMM (fast path) ==========================
// Y[m,n] = act( bn( sum_k X[row(m),k] * Wp[n,k] + bias[n] ) + residual[m,n] )
// X row-major f32 [*,K]; Wp packed f16 [N,K]; K % 32 == 0; X/Wp 256B aligned.
// rows != null gathers A rows. One wave computes a 16x32 tile of Y.
// Edge tiles: A-row / B-col indices are CLAMPED (garbage only affects rows /
// cols that are never stored), so all loads are unconditional and vectorized.
__global__ __launch_bounds__(THREADS)
void gemm_wmma_kernel(const float* __restrict__ X, const int* __restrict__ rows,
                      const _Float16* __restrict__ Wp,
                      const float* __restrict__ bias,
                      const float* __restrict__ gamma, const float* __restrict__ beta,
                      const float* __restrict__ residual,
                      float* __restrict__ Y,
                      int M, int N, int K, int do_relu)
{
    const int lane = threadIdx.x & 31;
    const int wave = threadIdx.x >> 5;
    const int mtile = (blockIdx.x * WAVES_PER_BLOCK + wave) * 16;
    const int ntile0 = blockIdx.y * 32;           // two adjacent 16-wide N tiles
    const int ntile1 = ntile0 + 16;
    if (mtile >= M) return;                       // wave-uniform exit

    const int l15    = lane & 15;
    const int mrowc  = min(mtile + l15, M - 1);   // clamped A row
    const int ncol0  = ntile0 + l15;              // for guarded stores
    const int ncol1  = ntile1 + l15;
    const int ncol0c = min(ncol0, N - 1);         // clamped B cols
    const int ncol1c = min(ncol1, N - 1);
    const long xrow  = rows ? (long)rows[mrowc] : (long)mrowc;

    // A 16x32 layout (matches B): lanes 0-15 hold K 0-15;
    //                             lanes 16-31 hold K 16-31.
    const int kA0 = (lane < 16) ? 0  : 16;
    const int kA1 = (lane < 16) ? 8  : 24;
    // B 32x16: lanes 0-15 hold K 0-15 (col=lane), lanes 16-31 hold K 16-31.
    const int kB0 = (lane < 16) ? 0 : 16;

    const float*    xp  = X  + xrow * (long)K;
    const _Float16* wp0 = Wp + (long)ncol0c * K;
    const _Float16* wp1 = Wp + (long)ncol1c * K;

    v8f acc0 = {};
    v8f acc1 = {};
    for (int k0 = 0; k0 < K; k0 += 32) {
        // ---- A fragment: 4 x float4 unconditional loads + cvt to f16 ----
        v4f x0 = *(const v4f*)(xp + k0 + kA0);
        v4f x1 = *(const v4f*)(xp + k0 + kA0 + 4);
        v4f x2 = *(const v4f*)(xp + k0 + kA1);
        v4f x3 = *(const v4f*)(xp + k0 + kA1 + 4);
        union { v16h v; _Float16 h[16]; } a;
#pragma unroll
        for (int i = 0; i < 4; ++i) {
            a.h[i]      = (_Float16)x0[i];
            a.h[4 + i]  = (_Float16)x1[i];
            a.h[8 + i]  = (_Float16)x2[i];
            a.h[12 + i] = (_Float16)x3[i];
        }
        // ---- B fragments: contiguous 32-byte f16 loads (one per N tile) ----
        v16h b0 = *(const v16h*)(wp0 + k0 + kB0);
        v16h b1 = *(const v16h*)(wp1 + k0 + kB0);

        acc0 = __builtin_amdgcn_wmma_f32_16x16x32_f16(false, a.v, false, b0,
                                                      (short)0, acc0, false, false);
        acc1 = __builtin_amdgcn_wmma_f32_16x16x32_f16(false, a.v, false, b1,
                                                      (short)0, acc1, false, false);
    }

    // fused epilogue:  +bias -> BN (g*rsqrt(1+eps), +beta) -> +residual -> relu
    const float bnscale = rsqrtf(1.0f + 1e-5f);
    float bi0 = 0.0f, sc0 = 1.0f, sh0 = 0.0f;
    float bi1 = 0.0f, sc1 = 1.0f, sh1 = 0.0f;
    if (bias)  { bi0 = bias[ncol0c]; bi1 = bias[ncol1c]; }
    if (gamma) { sc0 = gamma[ncol0c] * bnscale; sh0 = beta[ncol0c];
                 sc1 = gamma[ncol1c] * bnscale; sh1 = beta[ncol1c]; }
    const int rbase = (lane < 16) ? 0 : 8;
    union { v8f v; float f[8]; } c0, c1; c0.v = acc0; c1.v = acc1;
#pragma unroll
    for (int r = 0; r < 8; ++r) {
        int m = mtile + rbase + r;
        if (m < M) {
            if (ncol0 < N) {
                float val = c0.f[r] + bi0;
                if (gamma)    val = val * sc0 + sh0;
                if (residual) val += residual[(long)m * N + ncol0];
                if (do_relu)  val = fmaxf(val, 0.0f);
                Y[(long)m * N + ncol0] = val;
            }
            if (ncol1 < N) {
                float val = c1.f[r] + bi1;
                if (gamma)    val = val * sc1 + sh1;
                if (residual) val += residual[(long)m * N + ncol1];
                if (do_relu)  val = fmaxf(val, 0.0f);
                Y[(long)m * N + ncol1] = val;
            }
        }
    }
}

// ============== plain VALU linear for tiny K (stem K=4, pos K=3) ===========
__global__ void linear_smallk_kernel(const float* __restrict__ X, const float* __restrict__ W,
                                     const float* __restrict__ bias,
                                     const float* __restrict__ gamma, const float* __restrict__ beta,
                                     float* __restrict__ Y, long M, int N, int K, int do_relu)
{
    long t = (long)blockIdx.x * blockDim.x + threadIdx.x;
    if (t >= M * (long)N) return;
    long m = t / N; int n = (int)(t % N);
    const float* xp = X + m * K;
    const float* wpN = W + (long)n * K;
    float acc = bias ? bias[n] : 0.0f;
    for (int k = 0; k < K; ++k) acc += xp[k] * wpN[k];
    if (gamma) acc = acc * (gamma[n] * rsqrtf(1.0f + 1e-5f)) + beta[n];
    if (do_relu) acc = fmaxf(acc, 0.0f);
    Y[t] = acc;
}

// =============== weight packing: f32 [N,K] -> f16 [N,Ks] zero-padded =======
__global__ void pack_w_kernel(const float* __restrict__ W, _Float16* __restrict__ Wp,
                              int K, int Ks, long total /*N*Ks*/)
{
    long t = (long)blockIdx.x * blockDim.x + threadIdx.x;
    if (t >= total) return;
    long n = t / Ks; int k = (int)(t % Ks);
    Wp[t] = (k < K) ? (_Float16)W[n * (long)K + k] : (_Float16)0.0f;
}

// ========================== FPS (one block per batch) ======================
__global__ __launch_bounds__(THREADS)
void fps_kernel(const float* __restrict__ xyz, int* __restrict__ cents, int N, int S)
{
    __shared__ float dist[2048];
    __shared__ float rd[THREADS];
    __shared__ int   ri[THREADS];
    __shared__ int   s_far;
    const int b = blockIdx.x, tid = threadIdx.x;
    const float* p = xyz + (long)b * N * 3;

    for (int j = tid; j < N; j += THREADS) dist[j] = 1e10f;
    if (tid == 0) s_far = 0;
    __syncthreads();

    for (int it = 0; it < S; ++it) {
        int far = s_far;
        if (tid == 0) cents[(long)b * S + it] = far;     // emit pre-update (lax.scan)
        float cx = p[3 * far], cy = p[3 * far + 1], cz = p[3 * far + 2];
        float bestd = -1.0f; int besti = 0;
        for (int j = tid; j < N; j += THREADS) {
            float dx = p[3 * j] - cx, dy = p[3 * j + 1] - cy, dz = p[3 * j + 2] - cz;
            float nd = fminf(dist[j], dx * dx + dy * dy + dz * dz);
            dist[j] = nd;
            if (nd > bestd) { bestd = nd; besti = j; }   // strict > keeps first max
        }
        rd[tid] = bestd; ri[tid] = besti;
        __syncthreads();
        for (int s = THREADS / 2; s > 0; s >>= 1) {
            if (tid < s) {
                if (rd[tid + s] > rd[tid] ||
                    (rd[tid + s] == rd[tid] && ri[tid + s] < ri[tid])) {
                    rd[tid] = rd[tid + s]; ri[tid] = ri[tid + s];
                }
            }
            __syncthreads();
        }
        if (tid == 0) s_far = ri[0];
        __syncthreads();
    }
}

// ========================== KNN (16-NN insertion sort) =====================
__global__ __launch_bounds__(THREADS)
void knn_kernel(const float* __restrict__ xyz, const float* __restrict__ qry,
                int* __restrict__ idx, int N, int S, int total /*B*S*/)
{
    int t = blockIdx.x * blockDim.x + threadIdx.x;
    if (t >= total) return;
    int b = t / S;
    const float* q = qry + (long)t * 3;
    float qx = q[0], qy = q[1], qz = q[2];
    const float* p = xyz + (long)b * N * 3;
    float bd[16]; int bi[16];
#pragma unroll
    for (int i = 0; i < 16; ++i) { bd[i] = 3.4e38f; bi[i] = 0; }
    for (int j = 0; j < N; ++j) {
        float dx = p[3 * j] - qx, dy = p[3 * j + 1] - qy, dz = p[3 * j + 2] - qz;
        float d = dx * dx + dy * dy + dz * dz;
        if (d < bd[15]) {
            int pos = 15;
            while (pos > 0 && d < bd[pos - 1]) {
                bd[pos] = bd[pos - 1]; bi[pos] = bi[pos - 1]; --pos;
            }
            bd[pos] = d; bi[pos] = j;
        }
    }
    int* o = idx + (long)t * 16;
#pragma unroll
    for (int i = 0; i < 16; ++i) o[i] = bi[i];
}

// ========================== small elementwise kernels ======================
__global__ void extract_xyz_kernel(const float* __restrict__ x, float* __restrict__ xyz, int total)
{
    int t = blockIdx.x * blockDim.x + threadIdx.x;
    if (t >= total) return;
    xyz[3 * t] = x[4 * t]; xyz[3 * t + 1] = x[4 * t + 1]; xyz[3 * t + 2] = x[4 * t + 2];
}

__global__ void gather_points3_kernel(const float* __restrict__ xyz, const int* __restrict__ cents,
                                      float* __restrict__ out, int N, int S, int total /*B*S*/)
{
    int t = blockIdx.x * blockDim.x + threadIdx.x;
    if (t >= total) return;
    int b = t / S;
    const float* p = xyz + ((long)b * N + cents[t]) * 3;
    out[3 * t] = p[0]; out[3 * t + 1] = p[1]; out[3 * t + 2] = p[2];
}

// rel[b,s,j,:] = xyz[b, idx[b,s,j], :] - ctr[b,s,:]   (stride 3)
__global__ void rel_xyz_kernel(const float* __restrict__ xyz, const float* __restrict__ ctr,
                               const int* __restrict__ idx, float* __restrict__ rel,
                               int N, int S, int total /*B*S*16*/)
{
    int t = blockIdx.x * blockDim.x + threadIdx.x;
    if (t >= total) return;
    int bs = t >> 4;
    int b = bs / S;
    const float* p = xyz + ((long)b * N + idx[t]) * 3;
    const float* c = ctr + (long)bs * 3;
    rel[3 * t] = p[0] - c[0]; rel[3 * t + 1] = p[1] - c[1]; rel[3 * t + 2] = p[2] - c[2];
}

// rows[m] = b*N + idx[m]   (flat gather rows for indexed GEMM)
__global__ void rows_kernel(const int* __restrict__ idx, int* __restrict__ rows,
                            int N, int S, int total /*B*S*16*/)
{
    int t = blockIdx.x * blockDim.x + threadIdx.x;
    if (t >= total) return;
    int b = t / (S * 16);
    rows[t] = b * N + idx[t];
}

// kk <- q - kk + pos   (attn MLP input, in place over kk)
__global__ void attn_in_kernel(const float* __restrict__ q, float* __restrict__ kk,
                               const float* __restrict__ pos, int C, long total)
{
    long t = (long)blockIdx.x * blockDim.x + threadIdx.x;
    if (t >= total) return;
    long m = t / C; int c = (int)(t % C);
    kk[t] = q[(m >> 4) * C + c] - kk[t] + pos[t];
}

// softmax over the 16-neighbor axis, per (b,s,channel)
__global__ void softmax_nb_kernel(float* __restrict__ a, int C, int total /*B*S*C*/)
{
    int t = blockIdx.x * blockDim.x + threadIdx.x;
    if (t >= total) return;
    int c = t % C; long bs = t / C;
    float* p = a + (bs * 16) * (long)C + c;
    float mx = -3.4e38f;
    for (int j = 0; j < 16; ++j) mx = fmaxf(mx, p[(long)j * C]);
    float sum = 0.0f;
    for (int j = 0; j < 16; ++j) { float e = __expf(p[(long)j * C] - mx); p[(long)j * C] = e; sum += e; }
    float inv = 1.0f / sum;
    for (int j = 0; j < 16; ++j) p[(long)j * C] *= inv;
}

// out[bs,c] = sum_j attn * (v + pos)
__global__ void attn_sum_kernel(const float* __restrict__ attn, const float* __restrict__ v,
                                const float* __restrict__ pos, float* __restrict__ out,
                                int C, int total /*B*S*C*/)
{
    int t = blockIdx.x * blockDim.x + threadIdx.x;
    if (t >= total) return;
    int c = t % C; long bs = t / C;
    long base = (bs * 16) * (long)C + c;
    float s = 0.0f;
    for (int j = 0; j < 16; ++j) {
        long o = base + (long)j * C;
        s += attn[o] * (v[o] + pos[o]);
    }
    out[(long)t] = s;
}

// concat [gathered feat | gathered xyz - center | zero pad] -> [B,S,16,Ks]
__global__ void td_concat_kernel(const float* __restrict__ feat, const float* __restrict__ xyz,
                                 const float* __restrict__ ctr, const int* __restrict__ idx,
                                 float* __restrict__ out, int N, int S, int Ci, int Ks,
                                 int total /*B*S*16*/)
{
    int t = blockIdx.x * blockDim.x + threadIdx.x;
    if (t >= total) return;
    int bs = t >> 4;
    int b = bs / S;
    int src = idx[t];
    const float* f = feat + ((long)b * N + src) * Ci;
    float* o = out + (long)t * Ks;
    for (int c = 0; c < Ci; ++c) o[c] = f[c];
    const float* p = xyz + ((long)b * N + src) * 3;
    const float* cc = ctr + (long)bs * 3;
    o[Ci] = p[0] - cc[0]; o[Ci + 1] = p[1] - cc[1]; o[Ci + 2] = p[2] - cc[2];
    for (int c = Ci + 3; c < Ks; ++c) o[c] = 0.0f;   // zero pad to K%32==0
}

// max over the 16-neighbor axis
__global__ void maxpool16_kernel(const float* __restrict__ h, float* __restrict__ out,
                                 int C, int total /*B*S*C*/)
{
    int t = blockIdx.x * blockDim.x + threadIdx.x;
    if (t >= total) return;
    int c = t % C; long bs = t / C;
    long base = (bs * 16) * (long)C + c;
    float m = h[base];
    for (int j = 1; j < 16; ++j) m = fmaxf(m, h[base + (long)j * C]);
    out[(long)t] = m;
}

// max over the point axis
__global__ void maxpool_pts_kernel(const float* __restrict__ f, float* __restrict__ out,
                                   int S, int C, int total /*B*C*/)
{
    int t = blockIdx.x * blockDim.x + threadIdx.x;
    if (t >= total) return;
    int c = t % C; int b = t / C;
    const float* p = f + (long)b * S * C + c;
    float m = p[0];
    for (int j = 1; j < S; ++j) m = fmaxf(m, p[(long)j * C]);
    out[(long)t] = m;
}

// ========================== host-side orchestration ========================
struct BNp  { const float* g; const float* b; };
struct MLP2 { const float *W1, *W2, *b1, *b2; BNp bn; };
struct PTL  { MLP2 attn; const float* k; const float* out; MLP2 pos; const float* q; const float* v; };
struct PTB  { BNp bn1, bn2; const float *fc1, *fc2; PTL pt; };
struct TDp  { const float *W1, *W2; BNp bn1, bn2; };

struct Cursor {
    void* const* in; int i;
    const float* next() { return (const float*)in[i++]; }
};
// JAX pytree (sorted-dict-key) leaf order
static BNp  parseBN (Cursor& c){ BNp r; r.g = c.next(); r.b = c.next(); return r; }
static MLP2 parseMLP2(Cursor& c){ MLP2 m; m.W1=c.next(); m.W2=c.next(); m.b1=c.next(); m.b2=c.next(); m.bn=parseBN(c); return m; }
static PTL  parsePTL(Cursor& c){ PTL p; p.attn=parseMLP2(c); p.k=c.next(); p.out=c.next(); p.pos=parseMLP2(c); p.q=c.next(); p.v=c.next(); return p; }
static PTB  parsePTB(Cursor& c){ PTB p; p.bn1=parseBN(c); p.bn2=parseBN(c); p.fc1=c.next(); p.fc2=c.next(); p.pt=parsePTL(c); return p; }
static TDp  parseTD (Cursor& c){ TDp t; t.W1=c.next(); t.W2=c.next(); t.bn1=parseBN(c); t.bn2=parseBN(c); return t; }

struct Alloc {
    char* base; size_t off;
    float*    f(size_t n){ float*    p=(float*)   (base+off); off=(off+n*4+255)&~(size_t)255; return p; }
    int*      i(size_t n){ int*      p=(int*)     (base+off); off=(off+n*4+255)&~(size_t)255; return p; }
    _Float16* h(size_t n){ _Float16* p=(_Float16*)(base+off); off=(off+n*2+255)&~(size_t)255; return p; }
};

static inline dim3 grid1(long total){ return dim3((unsigned)((total + THREADS - 1) / THREADS)); }

static void gemm(hipStream_t s, const float* X, const int* rows, const _Float16* Wp,
                 const float* bias, const BNp* bn, const float* residual,
                 float* Y, int M, int N, int K, bool relu)
{
    dim3 g((M + 16 * WAVES_PER_BLOCK - 1) / (16 * WAVES_PER_BLOCK), (N + 31) / 32);
    gemm_wmma_kernel<<<g, THREADS, 0, s>>>(X, rows, Wp, bias,
                                           bn ? bn->g : nullptr, bn ? bn->b : nullptr,
                                           residual, Y, M, N, K, relu ? 1 : 0);
}

static void linear_smallk(hipStream_t s, const float* X, const float* W,
                          const float* bias, const BNp* bn,
                          float* Y, long M, int N, int K, bool relu)
{
    linear_smallk_kernel<<<grid1(M * N), THREADS, 0, s>>>(X, W, bias,
                                                          bn ? bn->g : nullptr, bn ? bn->b : nullptr,
                                                          Y, M, N, K, relu ? 1 : 0);
}

static const _Float16* packW(hipStream_t s, Alloc& a, const float* W, int N, int K, int Ks)
{
    _Float16* p = a.h((size_t)N * Ks);
    long total = (long)N * Ks;
    pack_w_kernel<<<grid1(total), THREADS, 0, s>>>(W, p, K, Ks, total);
    return p;
}

struct PackedPTB { const _Float16 *fc1, *fc2, *q, *k, *v, *out, *posW2, *attnW1, *attnW2; };
static PackedPTB packPTB(hipStream_t s, Alloc& a, const PTB& p, int C)
{
    PackedPTB w;
    w.fc1    = packW(s, a, p.fc1,        C, C, C);
    w.fc2    = packW(s, a, p.fc2,        C, C, C);
    w.q      = packW(s, a, p.pt.q,       C, C, C);
    w.k      = packW(s, a, p.pt.k,       C, C, C);
    w.v      = packW(s, a, p.pt.v,       C, C, C);
    w.out    = packW(s, a, p.pt.out,     C, C, C);
    w.posW2  = packW(s, a, p.pt.pos.W2,  C, C, C);
    w.attnW1 = packW(s, a, p.pt.attn.W1, C, C, C);
    w.attnW2 = packW(s, a, p.pt.attn.W2, C, C, C);
    return w;
}

struct PackedTD { const _Float16 *W1, *W2; int K1s; };
static PackedTD packTD(hipStream_t s, Alloc& a, const TDp& p, int Ci, int Co)
{
    PackedTD w;
    w.K1s = (Ci + 3 + 31) & ~31;
    w.W1 = packW(s, a, p.W1, Co, Ci + 3, w.K1s);
    w.W2 = packW(s, a, p.W2, Co, Co, Co);
    return w;
}

static const int Bsz = 16;

static void run_pt_block(hipStream_t s, const PTB& p, const PackedPTB& w,
                         const float* xyz, float* feat, int Np, int C, Alloc scratch)
{
    const int  M   = Bsz * Np;
    const int  M16 = M * 16;
    float* xb   = scratch.f((size_t)M * C);
    int*   idx  = scratch.i((size_t)M16);
    int*   rows = scratch.i((size_t)M16);
    float* rel  = scratch.f((size_t)M16 * 3);
    float* q    = scratch.f((size_t)M * C);
    float* kk   = scratch.f((size_t)M16 * C);
    float* v    = scratch.f((size_t)M16 * C);
    float* ph   = scratch.f((size_t)M16 * C);   // pos hidden, reused for attn hidden
    float* pos  = scratch.f((size_t)M16 * C);
    float* osum = scratch.f((size_t)M * C);
    float* oprj = scratch.f((size_t)M * C);

    gemm(s, feat, nullptr, w.fc1, nullptr, &p.bn1, nullptr, xb, M, C, C, true);
    knn_kernel<<<grid1(M), THREADS, 0, s>>>(xyz, xyz, idx, Np, Np, M);
    rows_kernel<<<grid1(M16), THREADS, 0, s>>>(idx, rows, Np, Np, M16);
    rel_xyz_kernel<<<grid1(M16), THREADS, 0, s>>>(xyz, xyz, idx, rel, Np, Np, M16);

    gemm(s, xb, nullptr, w.q, nullptr, nullptr, nullptr, q, M, C, C, false);
    gemm(s, xb, rows,    w.k, nullptr, nullptr, nullptr, kk, M16, C, C, false); // gathered
    gemm(s, xb, rows,    w.v, nullptr, nullptr, nullptr, v,  M16, C, C, false); // gathered

    // pos MLP layer1 has K=3 -> VALU kernel; layer2 K=C -> WMMA
    linear_smallk(s, rel, p.pt.pos.W1, p.pt.pos.b1, &p.pt.pos.bn, ph, M16, C, 3, true);
    gemm(s, ph, nullptr, w.posW2, p.pt.pos.b2, nullptr, nullptr, pos, M16, C, C, false);

    attn_in_kernel<<<grid1((long)M16 * C), THREADS, 0, s>>>(q, kk, pos, C, (long)M16 * C);
    gemm(s, kk, nullptr, w.attnW1, p.pt.attn.b1, &p.pt.attn.bn, nullptr, ph, M16, C, C, true);
    gemm(s, ph, nullptr, w.attnW2, p.pt.attn.b2, nullptr,       nullptr, kk, M16, C, C, false);
    softmax_nb_kernel<<<grid1((long)M * C), THREADS, 0, s>>>(kk, C, M * C);
    attn_sum_kernel<<<grid1((long)M * C), THREADS, 0, s>>>(kk, v, pos, osum, C, M * C);

    gemm(s, osum, nullptr, w.out, nullptr, nullptr, nullptr, oprj, M, C, C, false);
    // feat = relu(bn2(oprj @ fc2^T) + feat)  (in place; each element RMW by one lane)
    gemm(s, oprj, nullptr, w.fc2, nullptr, &p.bn2, feat, feat, M, C, C, true);
}

static void run_td(hipStream_t s, const TDp& p, const PackedTD& w,
                   const float* xyz, const float* feat,
                   int Np, int S, int Ci, int Co, float* new_xyz, float* new_feat, Alloc scratch)
{
    const int MS   = Bsz * S;
    const int MS16 = MS * 16;
    int*   cents = scratch.i((size_t)MS);
    int*   idx   = scratch.i((size_t)MS16);
    float* cat   = scratch.f((size_t)MS16 * w.K1s);
    float* h1    = scratch.f((size_t)MS16 * Co);
    float* h2    = scratch.f((size_t)MS16 * Co);

    fps_kernel<<<Bsz, THREADS, 0, s>>>(xyz, cents, Np, S);
    gather_points3_kernel<<<grid1(MS), THREADS, 0, s>>>(xyz, cents, new_xyz, Np, S, MS);
    knn_kernel<<<grid1(MS), THREADS, 0, s>>>(xyz, new_xyz, idx, Np, S, MS);
    td_concat_kernel<<<grid1(MS16), THREADS, 0, s>>>(feat, xyz, new_xyz, idx, cat,
                                                     Np, S, Ci, w.K1s, MS16);
    gemm(s, cat, nullptr, w.W1, nullptr, &p.bn1, nullptr, h1, MS16, Co, w.K1s, true);
    gemm(s, h1,  nullptr, w.W2, nullptr, &p.bn2, nullptr, h2, MS16, Co, Co, true);
    maxpool16_kernel<<<grid1((long)MS * Co), THREADS, 0, s>>>(h2, new_feat, Co, MS * Co);
}

extern "C" void kernel_launch(void* const* d_in, const int* in_sizes, int n_in,
                              void* d_out, int out_size, void* d_ws, size_t ws_size,
                              hipStream_t stream)
{
    (void)in_sizes; (void)n_in; (void)out_size; (void)ws_size;
    const float* x = (const float*)d_in[0];

    // params in pytree (alphabetical) order:
    // b1,b2,b3,b4, headW1, headW2, head_bn, headb2, stem_W, stem_bn, td1, td2, td3
    Cursor cur{d_in, 1};
    PTB b1 = parsePTB(cur), b2 = parsePTB(cur), b3 = parsePTB(cur), b4 = parsePTB(cur);
    const float* headW1 = cur.next();
    const float* headW2 = cur.next();
    BNp head_bn = parseBN(cur);
    const float* headb2 = cur.next();
    const float* stem_W = cur.next();
    BNp stem_bn = parseBN(cur);
    TDp td1 = parseTD(cur), td2 = parseTD(cur), td3 = parseTD(cur);

    const int N0 = 2048;
    Alloc ws{(char*)d_ws, 0};
    float* xyz0  = ws.f((size_t)Bsz * N0 * 3);
    float* feat0 = ws.f((size_t)Bsz * N0 * 32);
    float* xyz1  = ws.f((size_t)Bsz * 512 * 3);
    float* f1    = ws.f((size_t)Bsz * 512 * 64);
    float* xyz2  = ws.f((size_t)Bsz * 128 * 3);
    float* f2    = ws.f((size_t)Bsz * 128 * 128);
    float* xyz3  = ws.f((size_t)Bsz * 32 * 3);
    float* f3    = ws.f((size_t)Bsz * 32 * 256);
    float* gmax  = ws.f((size_t)Bsz * 256);
    float* hid   = ws.f((size_t)Bsz * 256);

    // pack all WMMA-path weights to f16 (zero-padded to K%32==0)
    PackedPTB wb1 = packPTB(stream, ws, b1, 32);
    PackedPTB wb2 = packPTB(stream, ws, b2, 64);
    PackedPTB wb3 = packPTB(stream, ws, b3, 128);
    PackedPTB wb4 = packPTB(stream, ws, b4, 256);
    PackedTD  wt1 = packTD(stream, ws, td1, 32, 64);
    PackedTD  wt2 = packTD(stream, ws, td2, 64, 128);
    PackedTD  wt3 = packTD(stream, ws, td3, 128, 256);
    const _Float16* wh1 = packW(stream, ws, headW1, 256, 256, 256);
    const _Float16* wh2 = packW(stream, ws, headW2, 26, 256, 256);

    Alloc scratch = ws;   // everything below is reused by each stage

    // stem: feat = relu(bn(x @ stem_W^T))  (K=4 -> VALU kernel, x rows stride 4)
    extract_xyz_kernel<<<grid1((long)Bsz * N0), THREADS, 0, stream>>>(x, xyz0, Bsz * N0);
    linear_smallk(stream, x, stem_W, nullptr, &stem_bn, feat0, (long)Bsz * N0, 32, 4, true);

    run_pt_block(stream, b1, wb1, xyz0, feat0, N0, 32, scratch);
    run_td(stream, td1, wt1, xyz0, feat0, N0, 512, 32, 64, xyz1, f1, scratch);
    run_pt_block(stream, b2, wb2, xyz1, f1, 512, 64, scratch);
    run_td(stream, td2, wt2, xyz1, f1, 512, 128, 64, 128, xyz2, f2, scratch);
    run_pt_block(stream, b3, wb3, xyz2, f2, 128, 128, scratch);
    run_td(stream, td3, wt3, xyz2, f2, 128, 32, 128, 256, xyz3, f3, scratch);
    run_pt_block(stream, b4, wb4, xyz3, f3, 32, 256, scratch);

    maxpool_pts_kernel<<<grid1((long)Bsz * 256), THREADS, 0, stream>>>(f3, gmax, 32, 256, Bsz * 256);
    gemm(stream, gmax, nullptr, wh1, nullptr, &head_bn, nullptr, hid, Bsz, 256, 256, true);
    gemm(stream, hid, nullptr, wh2, headb2, nullptr, nullptr, (float*)d_out, Bsz, 26, 256, false);
}
